// VectorGauntTensorProductAllParitiesS2Grid_11209864642663
// MI455X (gfx1250) — matrix operand order repaired and checked by reading, not compiled, not run on hardware
//
#include <hip/hip_runtime.h>
#include <hip/hip_bf16.h>

// ---------------------------------------------------------------------------
// VectorGauntTensorProductAllParitiesS2Grid for MI455X (gfx1250)
//  - f16 WMMA (v_wmma_f32_16x16x32_f16) for all GEMM stages, f32 accum
//  - Tensor Data Mover (tensor_load_to_lds) double-buffered staging of Y/Yw
//  - LDS transpose between synthesis (D-layout) and analysis (A-layout)
// ---------------------------------------------------------------------------

#define N_PTS 512
#define CIN 32
#define CC 32
#define LMAXC 4
#define DDIM 25
#define DPAD 32
#define RBQ 48
#define RAQ 95
#define NBB 3
#define G_RAW (RBQ * RAQ)   // 4560
#define G_PAD 4608          // multiple of 64
#define CHUNK 64
#define NCHUNK (G_PAD / CHUNK)  // 72
#define ROWS (N_PTS * CC)       // 16384
#define OUT_B 832
#define OUT_STRIDE (NBB * OUT_B) // 2496
#define PI_D 3.14159265358979323846

typedef __attribute__((ext_vector_type(16))) _Float16 v16h;
typedef __attribute__((ext_vector_type(8)))  _Float16 v8h;
typedef __attribute__((ext_vector_type(8)))  float    v8f;
typedef __attribute__((ext_vector_type(4)))  unsigned int u32x4;
typedef __attribute__((ext_vector_type(8)))  int      i32x8;
typedef __attribute__((ext_vector_type(4)))  int      i32x4;

// ---- workspace layout (bytes) ----
#define OFF_YT   0u                          // f16 [G_PAD][DPAD]
#define OFF_YW   294912u                     // f16 [DPAD][G_PAD]
#define OFF_C1   589824u                     // f16 [NBB][ROWS][DPAD]
#define OFF_C2   3735552u
#define OFF_CP   6881280u                    // f32 [NBB][N][DPAD][CC]
#define OFF_S    13172736u                   // f32 [N][CC]
#define OFF_W1T  13238272u                   // f16 [NBB][LMAX][CC][CIN]
#define OFF_W2T  13262848u
#define OFF_W0T  13287424u                   // f16 [NBB][64][64]
#define OFF_WLT  13312000u                   // f16 [NBB][LMAX][CC][CC]

// ---------------------------------------------------------------------------
__device__ __forceinline__ v8f wmma16(v16h a, v16h b, v8f c) {
  return __builtin_amdgcn_wmma_f32_16x16x32_f16(false, a, false, b,
                                                (short)0, c, false, false);
}

// 16 contiguous halves -> v16h (B fragment: lane has 16 consecutive K)
__device__ __forceinline__ v16h ldg16h(const _Float16* p) {
  const v8h a = *(const v8h*)p;
  const v8h b = *(const v8h*)(p + 8);
  v16h r;
#pragma unroll
  for (int i = 0; i < 8; ++i) { r[i] = a[i]; r[i + 8] = b[i]; }
  return r;
}

// A fragment (f16 16x32): lane kh half -> K {16kh..16kh+15}
__device__ __forceinline__ v16h ld_a16(const _Float16* row, int kh) {
  const v8h a = *(const v8h*)(row + 16 * kh);
  const v8h b = *(const v8h*)(row + 16 * kh + 8);
  v16h r;
#pragma unroll
  for (int i = 0; i < 8; ++i) { r[i] = a[i]; r[i + 8] = b[i]; }
  return r;
}

// A fragment built from f32 memory with cvt
__device__ __forceinline__ v16h ld_a32cvt(const float* row, int kh) {
  const float* p0 = row + 16 * kh;
  const float* p1 = row + 16 * kh + 8;
  v16h r;
#pragma unroll
  for (int i = 0; i < 8; ++i) {
    r[i]     = (_Float16)p0[i];
    r[i + 8] = (_Float16)p1[i];
  }
  return r;
}

__device__ __forceinline__ void decode_l(int r, int& l, int& t) {
  if (r < 96)       { l = 1; t = r; }
  else if (r < 256) { l = 2; t = r - 96; }
  else if (r < 480) { l = 3; t = r - 256; }
  else              { l = 4; t = r - 480; }
}

__device__ __forceinline__ unsigned lds_off(const void* p) {
  return (unsigned)(unsigned long long)p;   // LDS flat addr low 32 bits = offset
}

// Issue one TDM 2D tile load global->LDS (D# built from uniform scalars).
__device__ __forceinline__ void tdm_load(unsigned lds_addr,
                                         unsigned long long gaddr,
                                         unsigned data_size_log,
                                         unsigned tensor_d0, unsigned tensor_d1,
                                         unsigned tile_d0, unsigned tile_d1,
                                         unsigned long long stride0) {
  u32x4 g0;
  g0.x = 1u;                                       // count=1, load descriptor
  g0.y = lds_addr;                                 // LDS byte address
  g0.z = (unsigned)(gaddr & 0xFFFFFFFFull);        // global addr [31:0]
  g0.w = (unsigned)((gaddr >> 32) & 0x01FFFFFFull) | (2u << 30); // + type=2
  i32x8 g1;
  g1[0] = (int)(data_size_log << 16);              // workgroup_mask=0, data_size
  g1[1] = (int)((tensor_d0 & 0xFFFFu) << 16);      // tensor_dim0[15:0]
  g1[2] = (int)((tensor_d0 >> 16) | ((tensor_d1 & 0xFFFFu) << 16));
  g1[3] = (int)((tensor_d1 >> 16) | (tile_d0 << 16));
  g1[4] = (int)(tile_d1 & 0xFFFFu);                // tile_dim1, tile_dim2=0
  g1[5] = (int)(stride0 & 0xFFFFFFFFull);
  g1[6] = (int)((stride0 >> 32) & 0xFFFFull);
  g1[7] = 0;
  i32x4 z4 = {0, 0, 0, 0};
#if __clang_major__ >= 23
  i32x8 z8 = {0, 0, 0, 0, 0, 0, 0, 0};
  __builtin_amdgcn_tensor_load_to_lds(g0, g1, z4, z4, z8, 0);
#else
  __builtin_amdgcn_tensor_load_to_lds(g0, g1, z4, z4, 0);
#endif
}

// ---------------------------------------------------------------------------
// Kernel 1: build YT (f16 [G_PAD][32], B-layout for synthesis) and
//           Yw  (f16 [32][G_PAD], B-layout for analysis) on device.
// ---------------------------------------------------------------------------
__global__ void init_y_kernel(_Float16* __restrict__ YT,
                              _Float16* __restrict__ Ywh) {
  int g = blockIdx.x * blockDim.x + threadIdx.x;
  if (g >= G_PAD) return;
  if (g >= G_RAW) {
    for (int d = 0; d < DPAD; ++d) {
      YT[g * DPAD + d] = (_Float16)0.f;
      Ywh[d * G_PAD + g] = (_Float16)0.f;
    }
    return;
  }
  int qb = g / RAQ, ia = g % RAQ;
  // Gauss-Legendre node qb via Newton (double)
  double x = cos(PI_D * (qb + 0.75) / (RBQ + 0.5));
  for (int it = 0; it < 60; ++it) {
    double a = 1.0, b = x;
    for (int k = 2; k <= RBQ; ++k) {
      double c = ((2.0 * k - 1.0) * x * b - (k - 1.0) * a) / k;
      a = b; b = c;
    }
    double dp = RBQ * (x * b - a) / (x * x - 1.0);
    x -= b / dp;
  }
  double p1, dp;
  {
    double a = 1.0, b = x;
    for (int k = 2; k <= RBQ; ++k) {
      double c = ((2.0 * k - 1.0) * x * b - (k - 1.0) * a) / k;
      a = b; b = c;
    }
    p1 = b; dp = RBQ * (x * b - a) / (x * x - 1.0);
  }
  double wq = 2.0 / ((1.0 - x * x) * dp * dp);
  // associated Legendre (Condon-Shortley)
  double P[LMAXC + 1][LMAXC + 1];
  for (int l = 0; l <= LMAXC; ++l)
    for (int m = 0; m <= LMAXC; ++m) P[l][m] = 0.0;
  P[0][0] = 1.0;
  double sx = sqrt(fmax(0.0, 1.0 - x * x));
  for (int m = 1; m <= LMAXC; ++m) P[m][m] = -(2.0 * m - 1.0) * sx * P[m - 1][m - 1];
  for (int m = 0; m < LMAXC; ++m) P[m + 1][m] = (2.0 * m + 1.0) * x * P[m][m];
  for (int l = 2; l <= LMAXC; ++l)
    for (int m = 0; m <= l - 2; ++m)
      P[l][m] = ((2.0 * l - 1.0) * x * P[l - 1][m] - (l + m - 1.0) * P[l - 2][m]) / (l - m);
  const double fact[9] = {1, 1, 2, 6, 24, 120, 720, 5040, 40320};
  double alpha = 2.0 * PI_D * ia / RAQ;
  double wtot = wq * (2.0 * PI_D / RAQ);
  double Yv[DDIM];
  for (int l = 0; l <= LMAXC; ++l)
    for (int m = 0; m <= l; ++m) {
      double Nlm = sqrt((2.0 * l + 1.0) / (4.0 * PI_D) * fact[l - m] / fact[l + m]);
      if (m == 0) Yv[l * l + l] = Nlm * P[l][0];
      else {
        double base = sqrt(2.0) * Nlm * P[l][m];
        Yv[l * l + l + m] = base * cos(m * alpha);
        Yv[l * l + l - m] = base * sin(m * alpha);
      }
    }
  for (int d = 0; d < DPAD; ++d) {
    double yv = (d < DDIM) ? Yv[d] : 0.0;
    YT[g * DPAD + d] = (_Float16)(float)yv;
    Ywh[d * G_PAD + g] = (_Float16)(float)(yv * wtot);
  }
}

// ---------------------------------------------------------------------------
// Kernel 2: weight prep (transpose+cvt f16), s = x1[:,:,0]*x2[:,:,0],
//           zero-fill c1/c2 workspaces.
// ---------------------------------------------------------------------------
__global__ void prep_kernel(const float* __restrict__ x1, const float* __restrict__ x2,
                            const float* __restrict__ W1, const float* __restrict__ W2,
                            const float* __restrict__ Wout0, const float* __restrict__ WoutL,
                            _Float16* __restrict__ W1hT, _Float16* __restrict__ W2hT,
                            _Float16* __restrict__ W0hT, _Float16* __restrict__ WLhT,
                            float* __restrict__ s_ws,
                            _Float16* __restrict__ c1, _Float16* __restrict__ c2) {
  int tid = blockIdx.x * blockDim.x + threadIdx.x;
  int nth = gridDim.x * blockDim.x;
  // per-l weights: dest [bl][p][q] = src [bl][q][p]
  for (int i = tid; i < NBB * LMAXC * CC * CIN; i += nth) {
    int p = i & 31, q = (i >> 5) & 31, bl = i >> 10;
    W1hT[((long)bl * CC + p) * CIN + q] = (_Float16)W1[((long)bl * CIN + q) * CC + p];
    W2hT[((long)bl * CC + p) * CIN + q] = (_Float16)W2[((long)bl * CIN + q) * CC + p];
    WLhT[((long)bl * CC + p) * CC + q]  = (_Float16)WoutL[((long)bl * CC + q) * CC + p];
  }
  for (int i = tid; i < NBB * 64 * 64; i += nth) {
    int k = i & 63, col = (i >> 6) & 63, b = i >> 12;
    W0hT[((long)b * 64 + col) * 64 + k] = (_Float16)Wout0[((long)b * 64 + k) * 64 + col];
  }
  for (int i = tid; i < ROWS; i += nth)
    s_ws[i] = x1[(long)i * DDIM] * x2[(long)i * DDIM];
  for (int i = tid; i < NBB * ROWS * DPAD; i += nth) {
    c1[i] = (_Float16)0.f;
    c2[i] = (_Float16)0.f;
  }
}

// ---------------------------------------------------------------------------
// Kernel 3: equivariant projection via WMMA. Rows (n,m) grouped per l;
//           A = x[n,k,d] (K=cin), B = WhT[c][k]; scatter f16 into c_ws[row][d].
// ---------------------------------------------------------------------------
__global__ __launch_bounds__(256)
void proj_kernel(const float* __restrict__ x, const _Float16* __restrict__ WhT,
                 _Float16* __restrict__ cws) {
  int wid = threadIdx.x >> 5, lane = threadIdx.x & 31;
  int lane16 = lane & 15, kh = lane >> 4;
  int gw = blockIdx.x * 8 + wid;          // 0..2303
  int b = gw / 768, r = gw % 768;
  int l, t; decode_l(r, l, t);
  int ml = 2 * l + 1, d0 = l * l;
  int row = t * 16 + lane16;
  int n = row / ml, m = row % ml, d = d0 + m;
  v16h a;
#pragma unroll
  for (int j = 0; j < 8; ++j) {
    int k0 = 16 * kh + j, k1 = 16 * kh + 8 + j;
    a[j]     = (_Float16)x[((long)n * CIN + k0) * DDIM + d];
    a[j + 8] = (_Float16)x[((long)n * CIN + k1) * DDIM + d];
  }
  const _Float16* Wb = WhT + ((long)(b * LMAXC + (l - 1)) * CC) * CIN;
#pragma unroll
  for (int nt = 0; nt < 2; ++nt) {
    int c = nt * 16 + lane16;
    v16h bf = ldg16h(Wb + c * CIN + 16 * kh);
    v8f acc = {};
    acc = wmma16(a, bf, acc);
#pragma unroll
    for (int i = 0; i < 8; ++i) {
      int rr = t * 16 + i + 8 * kh;
      int nn = rr / ml, mm = rr % ml;
      cws[((long)b * ROWS + (long)nn * CC + c) * DPAD + d0 + mm] = (_Float16)acc[i];
    }
  }
}

// ---------------------------------------------------------------------------
// Kernel 4 (main): fused synthesis -> pointwise product -> analysis.
// One wave per 16-row (n,c) tile; Y/Yw chunks staged block-wide via TDM.
// ---------------------------------------------------------------------------
__global__ __launch_bounds__(256)
void gaunt_main_kernel(const _Float16* __restrict__ c1, const _Float16* __restrict__ c2,
                       const _Float16* __restrict__ YT, const _Float16* __restrict__ Ywh,
                       float* __restrict__ cp) {
  __shared__ __align__(16) _Float16 bufYT[2][CHUNK * DPAD]; // [g_local][k]
  __shared__ __align__(16) _Float16 bufYw[2][DPAD * CHUNK]; // [d][g_local]
  __shared__ __align__(16) _Float16 Pbuf[8][16 * CHUNK];    // [row][g_local] per wave
  int wid = threadIdx.x >> 5, lane = threadIdx.x & 31;
  int lane16 = lane & 15, kh = lane >> 4;
  int gw = blockIdx.x * 8 + wid;
  int b = gw >> 10, tile = gw & 1023;
  long rowbase = (long)b * ROWS + (long)tile * 16;

  v16h a1 = ld_a16(c1 + (rowbase + lane16) * DPAD, kh);
  v16h a2 = ld_a16(c2 + (rowbase + lane16) * DPAD, kh);
  v8f acc0 = {}, acc1 = {};
  bool issuer = (wid == 0);
  _Float16* pb = Pbuf[wid];

  if (issuer) {
#pragma unroll
    for (int pre = 0; pre < 2; ++pre) {
      // YT chunk: contiguous 4KB (1024 x 4B elements)
      tdm_load(lds_off(&bufYT[pre][0]),
               (unsigned long long)(const void*)(YT + (size_t)pre * CHUNK * DPAD),
               2u, 1024u, 1u, 1024u, 1u, 1024ull);
      // Yw chunk: 2D tile 64x32 of 2B elements, row stride G_PAD
      tdm_load(lds_off(&bufYw[pre][0]),
               (unsigned long long)(const void*)(Ywh + (size_t)pre * CHUNK),
               1u, (unsigned)G_PAD, 32u, (unsigned)CHUNK, 32u, (unsigned long long)G_PAD);
    }
  }

  for (int ch = 0; ch < NCHUNK; ++ch) {
    if (issuer) {
      if (ch + 1 < NCHUNK) __builtin_amdgcn_s_wait_tensorcnt(2);
      else                 __builtin_amdgcn_s_wait_tensorcnt(0);
    }
    __syncthreads();
    int bsel = ch & 1;
    const _Float16* yt = bufYT[bsel];
    const _Float16* yw = bufYw[bsel];
    // synthesis (S1 & S2 share the Y B-tile) + pointwise product -> LDS
#pragma unroll
    for (int nt = 0; nt < 4; ++nt) {
      v16h bf = ldg16h(yt + (nt * 16 + lane16) * DPAD + 16 * kh);
      v8f s1 = {}; s1 = wmma16(a1, bf, s1);
      v8f s2 = {}; s2 = wmma16(a2, bf, s2);
#pragma unroll
      for (int i = 0; i < 8; ++i)
        pb[(i + 8 * kh) * CHUNK + nt * 16 + lane16] = (_Float16)(s1[i] * s2[i]);
    }
    // analysis: CP += P @ Yw^T  (2 K-steps of 32, N-tiles d0..15 / d16..31)
#pragma unroll
    for (int ks = 0; ks < 2; ++ks) {
      v16h af = ld_a16(pb + lane16 * CHUNK + ks * 32, kh);
      v16h b0 = ldg16h(yw + lane16 * CHUNK + ks * 32 + 16 * kh);
      acc0 = wmma16(af, b0, acc0);
      v16h b1 = ldg16h(yw + (16 + lane16) * CHUNK + ks * 32 + 16 * kh);
      acc1 = wmma16(af, b1, acc1);
    }
    __syncthreads();
    if (issuer && ch + 2 < NCHUNK) {
      tdm_load(lds_off(&bufYT[bsel][0]),
               (unsigned long long)(const void*)(YT + (size_t)(ch + 2) * CHUNK * DPAD),
               2u, 1024u, 1u, 1024u, 1u, 1024ull);
      tdm_load(lds_off(&bufYw[bsel][0]),
               (unsigned long long)(const void*)(Ywh + (size_t)(ch + 2) * CHUNK),
               1u, (unsigned)G_PAD, 32u, (unsigned)CHUNK, 32u, (unsigned long long)G_PAD);
    }
  }

  // store CP to [b][n][d][c] (f32)
#pragma unroll
  for (int i = 0; i < 8; ++i) {
    long row = (long)tile * 16 + i + 8 * kh; // = n*32 + c
    int nn = (int)(row >> 5), cc2 = (int)(row & 31);
    cp[(((long)b * N_PTS + nn) * DPAD + lane16) * CC + cc2] = acc0[i];
    int dhi = 16 + lane16;
    if (dhi < DDIM)
      cp[(((long)b * N_PTS + nn) * DPAD + dhi) * CC + cc2] = acc1[i];
  }
}

// ---------------------------------------------------------------------------
// Kernel 5: o0 = [cp(:,:,0) | s] @ Wout0[b]   (K=64, N=64 via WMMA)
// ---------------------------------------------------------------------------
__global__ __launch_bounds__(256)
void out0_kernel(const float* __restrict__ cp, const float* __restrict__ s_ws,
                 const _Float16* __restrict__ W0hT, float* __restrict__ out) {
  int wid = threadIdx.x >> 5, lane = threadIdx.x & 31;
  int lane16 = lane & 15, kh = lane >> 4;
  int gw = blockIdx.x * 8 + wid;  // 0..95
  int b = gw / 32, t = gw % 32;
  int n = t * 16 + lane16;
  const float* cpr = cp + ((long)b * N_PTS + n) * DPAD * CC; // d=0 row, c contiguous
  v16h a0 = ld_a32cvt(cpr, kh);
  v16h a1 = ld_a32cvt(s_ws + (long)n * CC, kh);
#pragma unroll
  for (int nt = 0; nt < 4; ++nt) {
    int col = nt * 16 + lane16;
    const _Float16* wb = W0hT + ((long)b * 64 + col) * 64;
    v8f acc = {};
    acc = wmma16(a0, ldg16h(wb + 16 * kh), acc);
    acc = wmma16(a1, ldg16h(wb + 32 + 16 * kh), acc);
#pragma unroll
    for (int i = 0; i < 8; ++i) {
      int nn = t * 16 + i + 8 * kh;
      out[(long)nn * OUT_STRIDE + b * OUT_B + col] = acc[i];
    }
  }
}

// ---------------------------------------------------------------------------
// Kernel 6: oL = cp(:,:,l-block) contracted over c with WoutL[b,l] (WMMA)
// ---------------------------------------------------------------------------
__global__ __launch_bounds__(256)
void outL_kernel(const float* __restrict__ cp, const _Float16* __restrict__ WLhT,
                 float* __restrict__ out) {
  int wid = threadIdx.x >> 5, lane = threadIdx.x & 31;
  int lane16 = lane & 15, kh = lane >> 4;
  int gw = blockIdx.x * 8 + wid;  // 0..2303
  int b = gw / 768, r = gw % 768;
  int l, t; decode_l(r, l, t);
  int ml = 2 * l + 1, d0 = l * l;
  const int offl_tab[5] = {0, 64, 160, 320, 544};
  int offl = offl_tab[l];
  int row = t * 16 + lane16;
  int n = row / ml, m = row % ml;
  const float* cpr = cp + (((long)b * N_PTS + n) * DPAD + d0 + m) * CC;
  v16h a = ld_a32cvt(cpr, kh);
#pragma unroll
  for (int nt = 0; nt < 2; ++nt) {
    int dout = nt * 16 + lane16;
    v16h bf = ldg16h(WLhT + (((long)(b * LMAXC + (l - 1))) * CC + dout) * CC + 16 * kh);
    v8f acc = {};
    acc = wmma16(a, bf, acc);
#pragma unroll
    for (int i = 0; i < 8; ++i) {
      int rr = t * 16 + i + 8 * kh;
      int nn = rr / ml, mm = rr % ml;
      out[(long)nn * OUT_STRIDE + b * OUT_B + offl + dout * ml + mm] = acc[i];
    }
  }
}

// ---------------------------------------------------------------------------
extern "C" void kernel_launch(void* const* d_in, const int* in_sizes, int n_in,
                              void* d_out, int out_size, void* d_ws, size_t ws_size,
                              hipStream_t stream) {
  const float* x1    = (const float*)d_in[0];
  const float* x2    = (const float*)d_in[1];
  const float* W1    = (const float*)d_in[2];
  const float* W2    = (const float*)d_in[3];
  const float* Wout0 = (const float*)d_in[4];
  const float* WoutL = (const float*)d_in[5];
  float* out = (float*)d_out;
  char* ws = (char*)d_ws;

  _Float16* YT   = (_Float16*)(ws + OFF_YT);
  _Float16* Ywh  = (_Float16*)(ws + OFF_YW);
  _Float16* c1   = (_Float16*)(ws + OFF_C1);
  _Float16* c2   = (_Float16*)(ws + OFF_C2);
  float*    cp   = (float*)(ws + OFF_CP);
  float*    s_ws = (float*)(ws + OFF_S);
  _Float16* W1hT = (_Float16*)(ws + OFF_W1T);
  _Float16* W2hT = (_Float16*)(ws + OFF_W2T);
  _Float16* W0hT = (_Float16*)(ws + OFF_W0T);
  _Float16* WLhT = (_Float16*)(ws + OFF_WLT);

  init_y_kernel<<<(G_PAD + 255) / 256, 256, 0, stream>>>(YT, Ywh);
  prep_kernel<<<256, 256, 0, stream>>>(x1, x2, W1, W2, Wout0, WoutL,
                                       W1hT, W2hT, W0hT, WLhT, s_ws, c1, c2);
  proj_kernel<<<288, 256, 0, stream>>>(x1, W1hT, c1);
  proj_kernel<<<288, 256, 0, stream>>>(x2, W2hT, c2);
  gaunt_main_kernel<<<384, 256, 0, stream>>>(c1, c2, YT, Ywh, cp);
  out0_kernel<<<12, 256, 0, stream>>>(cp, s_ws, W0hT, out);
  outL_kernel<<<288, 256, 0, stream>>>(cp, WLhT, out);
}
